// Multi_CrossAttention_7722351198465
// MI455X (gfx1250) — matrix operand order repaired and checked by reading, not compiled, not run on hardware
//
#include <hip/hip_runtime.h>
#include <hip/hip_bf16.h>

typedef _Float16 h16v __attribute__((ext_vector_type(16)));
typedef _Float16 h8v  __attribute__((ext_vector_type(8)));
typedef _Float16 h4v  __attribute__((ext_vector_type(4)));
typedef float    f8v  __attribute__((ext_vector_type(8)));

union V16 { h16v v; h8v h[2]; };

#define BDIM 4
#define SDIM 2048
#define EDIM 512
#define HDIM 8
#define DDIM 64

// gfx1250 async memory->LDS copy (GVS mode). The instruction offset applies to
// both the global and the LDS destination address, so one LDS-address VGPR
// covers the +0/+16 byte pair.
#define ASYNC_B128(ldsoff, goff, base, imm)                                    \
  asm volatile("global_load_async_to_lds_b128 %0, %1, %2 offset:" #imm         \
               :: "v"(ldsoff), "v"(goff), "s"(base) : "memory")
#define S_WAIT_ASYNC(n) asm volatile("s_wait_asynccnt " #n ::: "memory")

static __device__ __forceinline__ f8v wmma16(const V16& a, const V16& b, f8v c) {
  // D = A(16x32 f16) * B(32x16 f16) + C(16x16 f32)
  return __builtin_amdgcn_wmma_f32_16x16x32_f16(false, a.v, false, b.v,
                                                (short)0, c, false, false);
}

static __device__ __forceinline__ float rowMax16(float v) {
  v = fmaxf(v, __shfl_xor(v, 1, 32));
  v = fmaxf(v, __shfl_xor(v, 2, 32));
  v = fmaxf(v, __shfl_xor(v, 4, 32));
  v = fmaxf(v, __shfl_xor(v, 8, 32));
  return v;
}
static __device__ __forceinline__ float rowSum16(float v) {
  v += __shfl_xor(v, 1, 32);
  v += __shfl_xor(v, 2, 32);
  v += __shfl_xor(v, 4, 32);
  v += __shfl_xor(v, 8, 32);
  return v;
}

// C[8192x512] = src[8192x512] * W[512x512]; store f16.
// transposed==0 -> dst layout [B,H,S,D]; transposed==1 -> dst layout [B,H,D,S]
__global__ __launch_bounds__(256) void proj_kernel(const float* __restrict__ src,
                                                   const float* __restrict__ W,
                                                   _Float16* __restrict__ dst,
                                                   int transposed) {
  __shared__ _Float16 As[128 * 32];  // [m][k]
  __shared__ _Float16 Bs[64 * 32];   // [n][k] (transposed so B-frag reads are contiguous)
  const int t = threadIdx.x;
  const int w = t >> 5, lane = t & 31, ln = lane & 15, hi = lane >> 4;
  const int mbase = blockIdx.x * 128;
  const int nbase = blockIdx.y * 64;

  f8v c[4] = {};
  for (int ks = 0; ks < 16; ++ks) {
    const int kbase = ks * 32;
    __syncthreads();
    // A tile 128x32 (f32 -> f16)
    for (int it = 0; it < 4; ++it) {
      const int lin = it * 1024 + t * 4;
      const int row = lin >> 5, col = lin & 31;
      const float4 a4 = *(const float4*)&src[(size_t)(mbase + row) * EDIM + kbase + col];
      h4v h;
      h[0] = (_Float16)a4.x; h[1] = (_Float16)a4.y;
      h[2] = (_Float16)a4.z; h[3] = (_Float16)a4.w;
      *(h4v*)&As[row * 32 + col] = h;
    }
    // B tile 32x64 -> Bs[n][k]
    for (int it = 0; it < 2; ++it) {
      const int lin = it * 1024 + t * 4;
      const int k = lin >> 6, n = lin & 63;
      const float4 b4 = *(const float4*)&W[(size_t)(kbase + k) * EDIM + nbase + n];
      Bs[(n + 0) * 32 + k] = (_Float16)b4.x;
      Bs[(n + 1) * 32 + k] = (_Float16)b4.y;
      Bs[(n + 2) * 32 + k] = (_Float16)b4.z;
      Bs[(n + 3) * 32 + k] = (_Float16)b4.w;
    }
    __syncthreads();

    V16 a;
    const int arow = w * 16 + ln;
    const int kb = hi * 8;
    a.h[0] = *(const h8v*)&As[arow * 32 + kb];
    a.h[1] = *(const h8v*)&As[arow * 32 + kb + 16];
    for (int nt = 0; nt < 4; ++nt) {
      V16 b;
      const int col = nt * 16 + ln;
      b.h[0] = *(const h8v*)&Bs[col * 32 + hi * 8];
      b.h[1] = *(const h8v*)&Bs[col * 32 + hi * 8 + 16];
      c[nt] = wmma16(a, b, c[nt]);
    }
  }

  for (int nt = 0; nt < 4; ++nt) {
    for (int r = 0; r < 8; ++r) {
      const int i = mbase + w * 16 + r + hi * 8;  // global row (b*S + s)
      const int b = i >> 11, s = i & (SDIM - 1);
      const int e = nbase + nt * 16 + ln;
      const int h = e >> 6, d = e & (DDIM - 1);
      const size_t off = transposed
          ? (((size_t)(b * HDIM + h) * DDIM + d) * SDIM + s)
          : (((size_t)(b * HDIM + h) * SDIM + s) * DDIM + d);
      dst[off] = (_Float16)c[nt][r];
    }
  }
}

// Fused dual-stream flash attention with double-buffered async K/V staging.
// Q1=y@Wq, K1=x@Wk, Vt1=(x@Wv)^T ; Q2=x@Wq, K2=y@Wk, Vt2=(y@Wv)^T
// attn = 0.5*(Q1 K1^T + Q2 K2^T)/sqrt(D); p=softmax; o1=p V1 + x; o2=p V2 + y
__global__ __launch_bounds__(256) void attn_kernel(
    const _Float16* __restrict__ Q1, const _Float16* __restrict__ K1,
    const _Float16* __restrict__ Vt1, const _Float16* __restrict__ Q2,
    const _Float16* __restrict__ K2, const _Float16* __restrict__ Vt2,
    const float* __restrict__ xin, const float* __restrict__ yin,
    float* __restrict__ out) {
  // [buffer][matrix: K1,K2,V1t,V2t][64x64 halves] = 64 KB, + 16 KB P staging
  __shared__ _Float16 KVbuf[2][4][64 * 64];
  __shared__ _Float16 Ps[8][16 * 64];

  const int t = threadIdx.x;
  const int w = t >> 5, lane = t & 31, ln = lane & 15, hi = lane >> 4;
  const int qbase = blockIdx.x * 128;
  const int bh = blockIdx.y;  // b*H + h
  const size_t headQK = (size_t)bh * SDIM * DDIM;
  const size_t headV  = (size_t)bh * DDIM * SDIM;
  const int kb = hi * 8;

  // Cooperative async staging: thread -> (row 0..63, 32-byte segment 0..3).
  const int row = t >> 2, seg = t & 3;
  const _Float16* kBase1 = K1 + headQK;
  const _Float16* kBase2 = K2 + headQK;
  const _Float16* vBase1 = Vt1 + headV;
  const _Float16* vBase2 = Vt2 + headV;
  auto issueTile = [&](int ktile, int buf) {
    const unsigned kOff = (unsigned)(((ktile * 64 + row) * DDIM) * 2 + seg * 32);
    const unsigned vOff = (unsigned)((row * SDIM + ktile * 64) * 2 + seg * 32);
    // low 32 bits of a generic LDS pointer == LDS byte offset (aperture rules)
    const unsigned lK1 = (unsigned)(size_t)&KVbuf[buf][0][row * 64 + seg * 16];
    const unsigned lK2 = (unsigned)(size_t)&KVbuf[buf][1][row * 64 + seg * 16];
    const unsigned lV1 = (unsigned)(size_t)&KVbuf[buf][2][row * 64 + seg * 16];
    const unsigned lV2 = (unsigned)(size_t)&KVbuf[buf][3][row * 64 + seg * 16];
    ASYNC_B128(lK1, kOff, kBase1, 0);  ASYNC_B128(lK1, kOff, kBase1, 16);
    ASYNC_B128(lK2, kOff, kBase2, 0);  ASYNC_B128(lK2, kOff, kBase2, 16);
    ASYNC_B128(lV1, vOff, vBase1, 0);  ASYNC_B128(lV1, vOff, vBase1, 16);
    ASYNC_B128(lV2, vOff, vBase2, 0);  ASYNC_B128(lV2, vOff, vBase2, 16);
  };

  // Q fragments (A-layout), 16 rows per wave, held for the whole key loop
  const int qrow = qbase + w * 16 + ln;
  V16 q1f[2], q2f[2];
  for (int ksl = 0; ksl < 2; ++ksl) {
    const size_t base = headQK + (size_t)qrow * DDIM + ksl * 32 + kb;
    q1f[ksl].h[0] = *(const h8v*)&Q1[base];
    q1f[ksl].h[1] = *(const h8v*)&Q1[base + 16];
    q2f[ksl].h[0] = *(const h8v*)&Q2[base];
    q2f[ksl].h[1] = *(const h8v*)&Q2[base + 16];
  }

  f8v o1[4] = {}, o2[4] = {};
  float mrun[8], lrun[8];
  for (int r = 0; r < 8; ++r) { mrun[r] = -1e30f; lrun[r] = 0.0f; }

  issueTile(0, 0);  // prologue: prefetch first tile (8 async ops/wave in flight)

  const int NT = SDIM / 64;  // 32 key tiles
  for (int kt = 0; kt < NT; ++kt) {
    const int cur = kt & 1;
    if (kt + 1 < NT) {
      issueTile(kt + 1, cur ^ 1);  // overlap: fetch next tile into other buffer
      S_WAIT_ASYNC(8);             // retire tile kt (older 8 ops; in-order)
    } else {
      S_WAIT_ASYNC(0);
    }
    __syncthreads();  // tile kt visible to all waves

    // S tile = Q1*K1^T + Q2*K2^T from LDS (shared across all 8 waves)
    f8v sacc[4] = {};
    for (int nt = 0; nt < 4; ++nt) {
      const int keyl = (nt * 16 + ln) * 64 + hi * 8;
      for (int ksl = 0; ksl < 2; ++ksl) {
        V16 bf;
        bf.h[0] = *(const h8v*)&KVbuf[cur][0][keyl + ksl * 32];
        bf.h[1] = *(const h8v*)&KVbuf[cur][0][keyl + ksl * 32 + 16];
        sacc[nt] = wmma16(q1f[ksl], bf, sacc[nt]);
        bf.h[0] = *(const h8v*)&KVbuf[cur][1][keyl + ksl * 32];
        bf.h[1] = *(const h8v*)&KVbuf[cur][1][keyl + ksl * 32 + 16];
        sacc[nt] = wmma16(q2f[ksl], bf, sacc[nt]);
      }
    }
    const float sc = 0.0625f;  // 0.5 (head average) / sqrt(64)
    for (int nt = 0; nt < 4; ++nt) sacc[nt] *= sc;

    // online softmax (row reductions across the 16-lane halves of C layout)
    float mnew[8], alpha[8], rs[8];
    for (int r = 0; r < 8; ++r) {
      float v = fmaxf(fmaxf(sacc[0][r], sacc[1][r]), fmaxf(sacc[2][r], sacc[3][r]));
      v = rowMax16(v);
      mnew[r] = fmaxf(mrun[r], v);
      alpha[r] = __expf(mrun[r] - mnew[r]);
      mrun[r] = mnew[r];
      rs[r] = 0.0f;
    }
    for (int nt = 0; nt < 4; ++nt)
      for (int r = 0; r < 8; ++r) {
        const float p = __expf(sacc[nt][r] - mnew[r]);
        rs[r] += p;
        Ps[w][(r + hi * 8) * 64 + nt * 16 + ln] = (_Float16)p;
      }
    for (int r = 0; r < 8; ++r) lrun[r] = lrun[r] * alpha[r] + rowSum16(rs[r]);
    for (int nt = 0; nt < 4; ++nt)
      for (int r = 0; r < 8; ++r) { o1[nt][r] *= alpha[r]; o2[nt][r] *= alpha[r]; }

    // P back as A-fragments (per-wave LDS region; DS ops in-order within wave)
    V16 pf[2];
    for (int ksl = 0; ksl < 2; ++ksl) {
      pf[ksl].h[0] = *(const h8v*)&Ps[w][ln * 64 + ksl * 32 + kb];
      pf[ksl].h[1] = *(const h8v*)&Ps[w][ln * 64 + ksl * 32 + kb + 16];
    }
    // O += P*V from LDS (V staged pre-transposed -> contiguous B-frags)
    for (int nt = 0; nt < 4; ++nt) {
      const int dl = (nt * 16 + ln) * 64 + hi * 8;
      for (int ksl = 0; ksl < 2; ++ksl) {
        V16 bf;
        bf.h[0] = *(const h8v*)&KVbuf[cur][2][dl + ksl * 32];
        bf.h[1] = *(const h8v*)&KVbuf[cur][2][dl + ksl * 32 + 16];
        o1[nt] = wmma16(pf[ksl], bf, o1[nt]);
        bf.h[0] = *(const h8v*)&KVbuf[cur][3][dl + ksl * 32];
        bf.h[1] = *(const h8v*)&KVbuf[cur][3][dl + ksl * 32 + 16];
        o2[nt] = wmma16(pf[ksl], bf, o2[nt]);
      }
    }
    __syncthreads();  // all waves done with buffer `cur` before it is refilled
  }

  // epilogue: normalize, add residual, write both outputs
  const int b = bh >> 3, h = bh & 7;
  for (int r = 0; r < 8; ++r) {
    const int q = qbase + w * 16 + r + hi * 8;
    const float invl = 1.0f / lrun[r];
    for (int nt = 0; nt < 4; ++nt) {
      const int e = h * 64 + nt * 16 + ln;
      const size_t oidx = ((size_t)b * SDIM + q) * EDIM + e;
      out[oidx] = o1[nt][r] * invl + xin[oidx];
      out[(size_t)BDIM * SDIM * EDIM + oidx] = o2[nt][r] * invl + yin[oidx];
    }
  }
}

extern "C" void kernel_launch(void* const* d_in, const int* in_sizes, int n_in,
                              void* d_out, int out_size, void* d_ws, size_t ws_size,
                              hipStream_t stream) {
  const float* x  = (const float*)d_in[0];
  const float* y  = (const float*)d_in[1];
  const float* Wq = (const float*)d_in[2];
  const float* Wk = (const float*)d_in[3];
  const float* Wv = (const float*)d_in[4];
  float* out = (float*)d_out;

  _Float16* ws = (_Float16*)d_ws;
  const size_t NQ = (size_t)BDIM * HDIM * SDIM * DDIM;  // 4,194,304 halves per slot

  _Float16* q1  = ws + 0 * NQ;  // y@Wq  [B,H,S,D]
  _Float16* k1  = ws + 1 * NQ;  // x@Wk  [B,H,S,D]
  _Float16* vt1 = ws + 2 * NQ;  // x@Wv  [B,H,D,S]
  _Float16* q2  = ws + 3 * NQ;  // x@Wq  [B,H,S,D]
  _Float16* k2  = ws + 4 * NQ;  // y@Wk  [B,H,S,D]
  _Float16* vt2 = ws + 5 * NQ;  // y@Wv  [B,H,D,S]

  const dim3 gProj(64, 8);  // 8192/128 x 512/64
  proj_kernel<<<gProj, 256, 0, stream>>>(y, Wq, q1, 0);
  proj_kernel<<<gProj, 256, 0, stream>>>(x, Wk, k1, 0);
  proj_kernel<<<gProj, 256, 0, stream>>>(x, Wv, vt1, 1);
  proj_kernel<<<gProj, 256, 0, stream>>>(x, Wq, q2, 0);
  proj_kernel<<<gProj, 256, 0, stream>>>(y, Wk, k2, 0);
  proj_kernel<<<gProj, 256, 0, stream>>>(y, Wv, vt2, 1);

  const dim3 gAttn(SDIM / 128, BDIM * HDIM);  // 16 x 32
  attn_kernel<<<gAttn, 256, 0, stream>>>(q1, k1, vt1, q2, k2, vt2, x, y, out);
}
